// cross_modal_module_69226282877400
// MI455X (gfx1250) — matrix operand order repaired and checked by reading, not compiled, not run on hardware
//
#include <hip/hip_runtime.h>
#include <stdint.h>

// ---------------- problem constants ----------------
#define L_     6
#define B_     16
#define N_     128
#define D_     512
#define KG_    50
#define KGP    64            // padded KG (multiple of 32 for WMMA K)
#define ROWS   (B_ * N_)     // 2048
#define KGROWS (B_ * N_ * N_)// 262144
#define FCW_LD (2 * D_ + KG_)// 1074
#define EPS_   1e-5f
#define SLOPE_ 0.01f

typedef __bf16 v16bf __attribute__((ext_vector_type(16)));
typedef __bf16 v8bf  __attribute__((ext_vector_type(8)));
typedef float  v8f   __attribute__((ext_vector_type(8)));
typedef int    v4i   __attribute__((ext_vector_type(4)));

union BF16x16 { v16bf v; v8bf h[2]; };

// ---------------- CDNA5 async global->LDS path (guarded) ----------------
#ifdef __has_builtin
#  if __has_builtin(__builtin_amdgcn_global_load_async_to_lds_b128) && \
      __has_builtin(__builtin_amdgcn_s_wait_asynccnt)
#    define USE_ASYNC_LDS 1
#  endif
#endif
#ifndef USE_ASYNC_LDS
#  define USE_ASYNC_LDS 0
#endif

#if USE_ASYNC_LDS
typedef __attribute__((address_space(1))) void as1v;
typedef __attribute__((address_space(3))) void as3v;
typedef __attribute__((address_space(1))) v4i  gv4i;
typedef __attribute__((address_space(3))) v4i  lv4i;
#endif

// XOR-swizzled LDS offset for a 128x128 bf16 tile: 16-byte segment index is
// XORed with (row & 15) so column-strided fragment reads are bank-conflict
// free while keeping 16B alignment for b128 accesses.
__device__ __forceinline__ int lds_off(int row, int col) {
  return row * 128 + ((((col >> 3) ^ (row & 15)) << 3) | (col & 7));
}

// Stage a 128x128 bf16 tile from global (leading dim ld) into swizzled LDS.
// 256 threads x 8 segments x 16B = 32KB.
__device__ __forceinline__ void lds_stage_128x128(const __bf16* __restrict__ g,
                                                  long ld, __bf16* lds, int tid) {
#pragma unroll
  for (int s = 0; s < 8; ++s) {
    const int seg = tid + 256 * s;           // 0..2047
    const int row = seg >> 4;
    const int col = (seg & 15) << 3;
    const __bf16* gp = g + (long)row * ld + col;
    __bf16* lp = lds + lds_off(row, col);
#if USE_ASYNC_LDS
    __builtin_amdgcn_global_load_async_to_lds_b128(
        (gv4i*)(as1v*)gp, (lv4i*)(as3v*)lp, 0, 0);
#else
    *(v8bf*)lp = *(const v8bf*)gp;
#endif
  }
}

__device__ __forceinline__ void lds_stage_wait() {
#if USE_ASYNC_LDS
  __builtin_amdgcn_s_wait_asynccnt(0);
#endif
  __syncthreads();
}

// ---------------- 64x32 WMMA macro-tile from swizzled LDS ----------------
// As: 128xKc A tile (use rows arow0..arow0+63); Bs: 128xKc B^T tile (rows =
// output columns, use brow0..brow0+31). 4 A-frags + 2 B-frags per K-step,
// 8 WMMAs.
__device__ __forceinline__ void wmma_64x32_lds(const __bf16* As, int arow0,
                                               const __bf16* Bs, int brow0,
                                               int Kc, int lane, v8f acc[4][2]) {
  const int mr = lane & 15;
  const int hi = lane >> 4;
  for (int k0 = 0; k0 < Kc; k0 += 32) {
    BF16x16 a[4], b[2];
#pragma unroll
    for (int i = 0; i < 4; ++i) {
      const int r = arow0 + i * 16 + mr;
      a[i].h[0] = *(const v8bf*)(As + lds_off(r, k0 + hi * 8));
      a[i].h[1] = *(const v8bf*)(As + lds_off(r, k0 + hi * 8 + 16));
    }
#pragma unroll
    for (int j = 0; j < 2; ++j) {
      const int r = brow0 + j * 16 + mr;
      b[j].h[0] = *(const v8bf*)(Bs + lds_off(r, k0 + hi * 8));
      b[j].h[1] = *(const v8bf*)(Bs + lds_off(r, k0 + hi * 8 + 16));
    }
#pragma unroll
    for (int i = 0; i < 4; ++i)
#pragma unroll
      for (int j = 0; j < 2; ++j)
        acc[i][j] = __builtin_amdgcn_wmma_f32_16x16x32_bf16(
            false, a[i].v, false, b[j].v, (short)0, acc[i][j], false, false);
  }
}

// ---------------- generic GEMM: C = act(A @ W + bias) (+ res) ----------------
// ACT: 0=none, 1=relu, 2=leaky-relu(0.01).
// Block computes a 128x128 tile (8 waves, each 64x32), K chunked by 128 via LDS.
template <int ACT, bool RES, bool ST32, bool STBF, bool STBFT>
__global__ __launch_bounds__(256) void gemm_kernel(
    const __bf16* __restrict__ A, const __bf16* __restrict__ Bt,
    const float* __restrict__ bias, const float* __restrict__ res,
    float* __restrict__ C32, __bf16* __restrict__ Cbf, __bf16* __restrict__ CbfT,
    int M, int N, int K) {
  __shared__ __bf16 As[128 * 128];
  __shared__ __bf16 Bs[128 * 128];
  const int tid = threadIdx.x;
  const int lane = tid & 31;
  const int wave = tid >> 5;
  const int NB = N >> 7;                 // 128-col blocks
  const int bm = blockIdx.x / NB, bn = blockIdx.x % NB;
  const int wm = wave >> 2, wn = wave & 3;   // 2 x 4 wave grid
  v8f acc[4][2] = {};
  for (int kc = 0; kc < K; kc += 128) {
    lds_stage_128x128(A + (long)bm * 128 * K + kc, K, As, tid);
    lds_stage_128x128(Bt + (long)bn * 128 * K + kc, K, Bs, tid);
    lds_stage_wait();
    wmma_64x32_lds(As, wm * 64, Bs, wn * 32, 128, lane, acc);
    __syncthreads();
  }
  const int nl = lane & 15, hi = lane >> 4;
#pragma unroll
  for (int i = 0; i < 4; ++i) {
#pragma unroll
    for (int j = 0; j < 2; ++j) {
      const int n = bn * 128 + wn * 32 + j * 16 + nl;
      const float bb = bias[n];
#pragma unroll
      for (int r = 0; r < 8; ++r) {
        const int m = bm * 128 + wm * 64 + i * 16 + r + 8 * hi;
        float x = acc[i][j][r] + bb;
        if (ACT == 1) x = fmaxf(x, 0.0f);
        if (ACT == 2) x = (x >= 0.0f) ? x : SLOPE_ * x;
        if (RES)   x += res[(long)m * N + n];
        if (ST32)  C32[(long)m * N + n] = x;
        if (STBF)  Cbf[(long)m * N + n] = (__bf16)x;
        if (STBFT) CbfT[(long)n * M + m] = (__bf16)x;  // transposed bf16 store
      }
    }
  }
}

// ---------------- attn @ v (+ v residual) -> src, per-batch block-diagonal ----
// One block per (batch, 128-col tile); K = 128 (one LDS chunk).
__global__ __launch_bounds__(256) void attn_v_kernel(
    const __bf16* __restrict__ attn,  // (B, 128, 128) bf16
    const __bf16* __restrict__ vT,    // (512, 2048) bf16 (v transposed)
    const float* __restrict__ v32,    // (2048, 512) f32
    float* __restrict__ src) {        // (2048, 512) f32
  __shared__ __bf16 As[128 * 128];
  __shared__ __bf16 Bs[128 * 128];
  const int tid = threadIdx.x;
  const int lane = tid & 31;
  const int wave = tid >> 5;
  const int bm = blockIdx.x >> 2;        // batch
  const int bn = blockIdx.x & 3;         // 128-col tile of D
  const int wm = wave >> 2, wn = wave & 3;
  lds_stage_128x128(attn + (long)bm * N_ * N_, N_, As, tid);
  lds_stage_128x128(vT + (long)bn * 128 * ROWS + bm * N_, ROWS, Bs, tid);
  lds_stage_wait();
  v8f acc[4][2] = {};
  wmma_64x32_lds(As, wm * 64, Bs, wn * 32, 128, lane, acc);
  const int nl = lane & 15, hi = lane >> 4;
#pragma unroll
  for (int i = 0; i < 4; ++i) {
#pragma unroll
    for (int j = 0; j < 2; ++j) {
      const int n = bn * 128 + wn * 32 + j * 16 + nl;
#pragma unroll
      for (int r = 0; r < 8; ++r) {
        const int m = bm * 128 + wm * 64 + i * 16 + r + 8 * hi;
        src[(long)m * D_ + n] = v32[(long)m * D_ + n] + acc[i][j][r];
      }
    }
  }
}

// ------- fused kg GEMM + relu + dot(fkg): skg[row] = sum_c relu(kg)*fkg[c] ----
__global__ __launch_bounds__(256) void kg_skg_kernel(
    const __bf16* __restrict__ KGb,   // (262144, 64) bf16 (padded)
    const __bf16* __restrict__ Wkgt,  // (64, 64) bf16 transposed+padded
    const float* __restrict__ bkg_p,  // (64,) padded
    const float* __restrict__ fkg_p,  // (64,) padded (0 beyond 50)
    float* __restrict__ skg) {        // (262144,)
  const int lane = threadIdx.x & 31;
  const int t = blockIdx.x * 8 + (threadIdx.x >> 5);  // 16-row block id
  const int mr = lane & 15, hi = lane >> 4;
  const __bf16* Arow = KGb + (long)t * 16 * KGP + mr * KGP + hi * 16;
  BF16x16 a0, a1;
  a0.h[0] = *(const v8bf*)(Arow + 0);   a0.h[1] = *(const v8bf*)(Arow + 8);
  a1.h[0] = *(const v8bf*)(Arow + 32);  a1.h[1] = *(const v8bf*)(Arow + 40);
  float p[8] = {0.f, 0.f, 0.f, 0.f, 0.f, 0.f, 0.f, 0.f};
#pragma unroll
  for (int ct = 0; ct < 4; ++ct) {
    const __bf16* Brow = Wkgt + (ct * 16 + mr) * KGP + hi * 16;
    BF16x16 b0, b1;
    b0.v = *(const v16bf*)(Brow);
    b1.v = *(const v16bf*)(Brow + 32);
    v8f acc = {};
    acc = __builtin_amdgcn_wmma_f32_16x16x32_bf16(false, a0.v, false, b0.v, (short)0, acc, false, false);
    acc = __builtin_amdgcn_wmma_f32_16x16x32_bf16(false, a1.v, false, b1.v, (short)0, acc, false, false);
    const int n = ct * 16 + mr;
    const float bb = bkg_p[n], ff = fkg_p[n];
#pragma unroll
    for (int r = 0; r < 8; ++r) {
      float x = acc[r] + bb;
      x = fmaxf(x, 0.0f);
      p[r] += x * ff;
    }
  }
#pragma unroll
  for (int m = 1; m < 16; m <<= 1)
#pragma unroll
    for (int r = 0; r < 8; ++r) p[r] += __shfl_xor(p[r], m, 16);
  if ((lane & 15) == 0) {
#pragma unroll
    for (int r = 0; r < 8; ++r) skg[(long)t * 16 + r + 8 * hi] = p[r];
  }
}

// ---------------- row dot: out[row] = X[row,:] . w ----------------
__global__ __launch_bounds__(256) void rowdot_kernel(
    const float* __restrict__ X, const float* __restrict__ w,
    float* __restrict__ out, int D) {
  const int lane = threadIdx.x & 31;
  const int row = blockIdx.x * 8 + (threadIdx.x >> 5);
  const float* x = X + (long)row * D;
  float s = 0.f;
  for (int j = lane; j < D; j += 32) s += x[j] * w[j];
#pragma unroll
  for (int m = 16; m >= 1; m >>= 1) s += __shfl_xor(s, m, 32);
  if (lane == 0) out[row] = s;
}

// ------- scores = lrelu(sq_i + sk_j + skg_ij + fb); softmax over j -> bf16 ----
__global__ __launch_bounds__(256) void softmax_attn_kernel(
    const float* __restrict__ sq, const float* __restrict__ sk,
    const float* __restrict__ skg, const float* __restrict__ fb_ptr,
    __bf16* __restrict__ attn) {
  const int lane = threadIdx.x & 31;
  const int rid = blockIdx.x * 8 + (threadIdx.x >> 5);  // (b,i) row, 0..2047
  const int b = rid >> 7;
  const float fb = fb_ptr[0];
  const float qv = sq[rid];
  float s[4];
  float mx = -1e30f;
#pragma unroll
  for (int t = 0; t < 4; ++t) {
    const int j = lane + 32 * t;
    float x = qv + sk[b * N_ + j] + skg[(long)rid * N_ + j] + fb;
    x = (x >= 0.0f) ? x : SLOPE_ * x;
    s[t] = x;
    mx = fmaxf(mx, x);
  }
#pragma unroll
  for (int m = 16; m >= 1; m >>= 1) mx = fmaxf(mx, __shfl_xor(mx, m, 32));
  float sum = 0.f;
#pragma unroll
  for (int t = 0; t < 4; ++t) { s[t] = __expf(s[t] - mx); sum += s[t]; }
#pragma unroll
  for (int m = 16; m >= 1; m >>= 1) sum += __shfl_xor(sum, m, 32);
  const float inv = 1.0f / sum;
#pragma unroll
  for (int t = 0; t < 4; ++t)
    attn[(long)rid * N_ + lane + 32 * t] = (__bf16)(s[t] * inv);
}

// ---------------- layernorm over D=512; output f32 or bf16 ----------------
template <bool OUTBF>
__global__ __launch_bounds__(256) void layernorm_kernel(
    const float* __restrict__ X, const float* __restrict__ g,
    const float* __restrict__ b, float* __restrict__ Y32,
    __bf16* __restrict__ Ybf) {
  const int lane = threadIdx.x & 31;
  const int row = blockIdx.x * 8 + (threadIdx.x >> 5);
  const float* x = X + (long)row * D_;
  float xl[16];
  float s = 0.f, s2 = 0.f;
#pragma unroll
  for (int t = 0; t < 16; ++t) {
    const float v = x[lane + 32 * t];
    xl[t] = v; s += v; s2 += v * v;
  }
#pragma unroll
  for (int m = 16; m >= 1; m >>= 1) { s += __shfl_xor(s, m, 32); s2 += __shfl_xor(s2, m, 32); }
  const float mean = s * (1.0f / D_);
  const float var = s2 * (1.0f / D_) - mean * mean;
  const float inv = rsqrtf(var + EPS_);
#pragma unroll
  for (int t = 0; t < 16; ++t) {
    const int j = lane + 32 * t;
    const float y = (xl[t] - mean) * inv * g[j] + b[j];
    if (OUTBF) Ybf[(long)row * D_ + j] = (__bf16)y;
    else       Y32[(long)row * D_ + j] = y;
  }
}

// ---------------- converts / pre-pass ----------------
__global__ void f32_to_bf16_kernel(const float* __restrict__ x,
                                   __bf16* __restrict__ y, int n) {
  const int i = blockIdx.x * blockDim.x + threadIdx.x;
  if (i < n) y[i] = (__bf16)x[i];
}

// W: (L, K, N) f32 -> Wt: (L, N, K) bf16
__global__ void wt_convert_kernel(const float* __restrict__ W,
                                  __bf16* __restrict__ Wt, int K, int N, int total) {
  const int i = blockIdx.x * blockDim.x + threadIdx.x;
  if (i >= total) return;
  const int l = i / (K * N);
  const int e = i % (K * N);
  const int n = e / K, k = e % K;
  Wt[i] = (__bf16)W[(long)l * K * N + (long)k * N + n];
}

// Wkg: (L,50,50) f32 -> (L,64,64) bf16 transposed + zero padded
__global__ void wkg_convert_kernel(const float* __restrict__ W,
                                   __bf16* __restrict__ Wt) {
  const int i = blockIdx.x * blockDim.x + threadIdx.x;
  if (i >= L_ * KGP * KGP) return;
  const int l = i / (KGP * KGP);
  const int e = i % (KGP * KGP);
  const int n = e / KGP, k = e % KGP;
  Wt[i] = (k < KG_ && n < KG_) ? (__bf16)W[(long)l * KG_ * KG_ + k * KG_ + n]
                               : (__bf16)0.0f;
}

// KG_matrix: (262144, 50) f32 -> (262144, 64) bf16 zero padded
__global__ void kg_convert_kernel(const float* __restrict__ KG,
                                  __bf16* __restrict__ KGb) {
  const long i = (long)blockIdx.x * blockDim.x + threadIdx.x;
  if (i >= (long)KGROWS * KGP) return;
  const long r = i / KGP;
  const int k = (int)(i % KGP);
  KGb[i] = (k < KG_) ? (__bf16)KG[r * KG_ + k] : (__bf16)0.0f;
}

// pad bkg (L,50) and fkg = fc_w[:,1024:1074] to (L,64) f32 (zeros beyond 50)
__global__ void pad_vec_kernel(const float* __restrict__ bkg,
                               const float* __restrict__ fcw,
                               float* __restrict__ bkg_p,
                               float* __restrict__ fkg_p) {
  const int i = blockIdx.x * blockDim.x + threadIdx.x;
  if (i >= L_ * KGP) return;
  const int l = i / KGP, c = i % KGP;
  bkg_p[i] = (c < KG_) ? bkg[l * KG_ + c] : 0.0f;
  fkg_p[i] = (c < KG_) ? fcw[l * FCW_LD + 2 * D_ + c] : 0.0f;
}

// ---------------- host orchestration ----------------
extern "C" void kernel_launch(void* const* d_in, const int* in_sizes, int n_in,
                              void* d_out, int out_size, void* d_ws, size_t ws_size,
                              hipStream_t stream) {
  const float* vectors = (const float*)d_in[0];
  const float* KG      = (const float*)d_in[1];
  const float* Wq = (const float*)d_in[2];
  const float* bq = (const float*)d_in[3];
  const float* Wk = (const float*)d_in[4];
  const float* bk = (const float*)d_in[5];
  const float* Wv = (const float*)d_in[6];
  const float* bv = (const float*)d_in[7];
  const float* Wkg= (const float*)d_in[8];
  const float* bkg= (const float*)d_in[9];
  const float* fcw= (const float*)d_in[10];
  const float* fcb= (const float*)d_in[11];
  const float* ng = (const float*)d_in[12];
  const float* nb = (const float*)d_in[13];
  const float* W1 = (const float*)d_in[14];
  const float* b1 = (const float*)d_in[15];
  const float* W2 = (const float*)d_in[16];
  const float* b2 = (const float*)d_in[17];
  const float* gf = (const float*)d_in[18];
  const float* bfv= (const float*)d_in[19];
  float* outp = (float*)d_out;

  // ---- workspace carve (256B aligned) ----
  char* base = (char*)d_ws;
  size_t off = 0;
  auto alloc = [&](size_t bytes) -> void* {
    off = (off + 255) & ~(size_t)255;
    void* r = (void*)(base + off);
    off += bytes;
    return r;
  };
  const size_t WT_BYTES = (size_t)L_ * D_ * D_ * 2;
  __bf16* Wt_q  = (__bf16*)alloc(WT_BYTES);
  __bf16* Wt_k  = (__bf16*)alloc(WT_BYTES);
  __bf16* Wt_v  = (__bf16*)alloc(WT_BYTES);
  __bf16* Wt_1  = (__bf16*)alloc(WT_BYTES);
  __bf16* Wt_2  = (__bf16*)alloc(WT_BYTES);
  __bf16* Wkgt  = (__bf16*)alloc((size_t)L_ * KGP * KGP * 2);
  __bf16* KGb   = (__bf16*)alloc((size_t)KGROWS * KGP * 2);
  float*  bkg_p = (float*)alloc((size_t)L_ * KGP * 4);
  float*  fkg_p = (float*)alloc((size_t)L_ * KGP * 4);
  float*  act   = (float*)alloc((size_t)ROWS * D_ * 4);
  __bf16* Abf   = (__bf16*)alloc((size_t)ROWS * D_ * 2);
  float*  q32   = (float*)alloc((size_t)ROWS * D_ * 4);
  float*  k32   = (float*)alloc((size_t)ROWS * D_ * 4);
  float*  v32   = (float*)alloc((size_t)ROWS * D_ * 4);
  __bf16* vT    = (__bf16*)alloc((size_t)D_ * ROWS * 2);
  float*  sq    = (float*)alloc((size_t)ROWS * 4);
  float*  sk    = (float*)alloc((size_t)ROWS * 4);
  float*  skg   = (float*)alloc((size_t)KGROWS * 4);
  __bf16* attn  = (__bf16*)alloc((size_t)KGROWS * 2);
  float*  src   = (float*)alloc((size_t)ROWS * D_ * 4);
  __bf16* s2bf  = (__bf16*)alloc((size_t)ROWS * D_ * 2);
  __bf16* hbf   = (__bf16*)alloc((size_t)ROWS * D_ * 2);
  if (off > ws_size) return;  // workspace too small; bail safely

  const dim3 blk(256);

  // ---- pre-pass: weight conversion (every call; deterministic) ----
  {
    const int tot = L_ * D_ * D_;
    const int g = (tot + 255) / 256;
    wt_convert_kernel<<<g, blk, 0, stream>>>(Wq, Wt_q, D_, D_, tot);
    wt_convert_kernel<<<g, blk, 0, stream>>>(Wk, Wt_k, D_, D_, tot);
    wt_convert_kernel<<<g, blk, 0, stream>>>(Wv, Wt_v, D_, D_, tot);
    wt_convert_kernel<<<g, blk, 0, stream>>>(W1, Wt_1, D_, D_, tot);
    wt_convert_kernel<<<g, blk, 0, stream>>>(W2, Wt_2, D_, D_, tot);
  }
  wkg_convert_kernel<<<(L_ * KGP * KGP + 255) / 256, blk, 0, stream>>>(Wkg, Wkgt);
  kg_convert_kernel<<<(int)(((long)KGROWS * KGP + 255) / 256), blk, 0, stream>>>(KG, KGb);
  pad_vec_kernel<<<(L_ * KGP + 255) / 256, blk, 0, stream>>>(bkg, fcw, bkg_p, fkg_p);
  (void)hipMemcpyAsync(act, vectors, (size_t)ROWS * D_ * 4,
                       hipMemcpyDeviceToDevice, stream);
  // initial act -> bf16 (subsequent layers fuse this into the output GEMM)
  f32_to_bf16_kernel<<<(ROWS * D_ + 255) / 256, blk, 0, stream>>>(act, Abf, ROWS * D_);

  const int GEMM_BLOCKS = (ROWS / 128) * (D_ / 128);    // 64 (128x128 tiles)
  const int ROWW_BLOCKS = ROWS / 8;                     // 256 (one wave per row)
  const int KG_BLOCKS   = (KGROWS / 16) / 8;            // 2048

  // ---- per-layer pipeline ----
  for (int l = 0; l < L_; ++l) {
    const __bf16* wtq = Wt_q + (size_t)l * D_ * D_;
    const __bf16* wtk = Wt_k + (size_t)l * D_ * D_;
    const __bf16* wtv = Wt_v + (size_t)l * D_ * D_;
    const __bf16* wt1 = Wt_1 + (size_t)l * D_ * D_;
    const __bf16* wt2 = Wt_2 + (size_t)l * D_ * D_;

    // q,k,v = relu(act @ W + b); v also stored transposed bf16
    gemm_kernel<1, false, true, false, false><<<GEMM_BLOCKS, blk, 0, stream>>>(
        Abf, wtq, bq + l * D_, nullptr, q32, nullptr, nullptr, ROWS, D_, D_);
    gemm_kernel<1, false, true, false, false><<<GEMM_BLOCKS, blk, 0, stream>>>(
        Abf, wtk, bk + l * D_, nullptr, k32, nullptr, nullptr, ROWS, D_, D_);
    gemm_kernel<1, false, true, false, true><<<GEMM_BLOCKS, blk, 0, stream>>>(
        Abf, wtv, bv + l * D_, nullptr, v32, nullptr, vT, ROWS, D_, D_);

    // sq = q . fw[:D], sk = k . fw[D:2D]
    rowdot_kernel<<<ROWW_BLOCKS, blk, 0, stream>>>(q32, fcw + (size_t)l * FCW_LD, sq, D_);
    rowdot_kernel<<<ROWW_BLOCKS, blk, 0, stream>>>(k32, fcw + (size_t)l * FCW_LD + D_, sk, D_);

    // skg[b,i,j] = relu(KG @ Wkg + bkg) . fw[2D:]
    kg_skg_kernel<<<KG_BLOCKS, blk, 0, stream>>>(
        KGb, Wkgt + (size_t)l * KGP * KGP, bkg_p + l * KGP, fkg_p + l * KGP, skg);

    // softmax(lrelu(sq_i + sk_j + skg_ij + fb)) -> bf16 attn
    softmax_attn_kernel<<<ROWW_BLOCKS, blk, 0, stream>>>(sq, sk, skg, fcb + l, attn);

    // src = v + attn @ v
    attn_v_kernel<<<GEMM_BLOCKS, blk, 0, stream>>>(attn, vT, v32, src);

    // src2 = LN(src) -> bf16
    layernorm_kernel<true><<<ROWW_BLOCKS, blk, 0, stream>>>(
        src, ng + l * D_, nb + l * D_, nullptr, s2bf);

    // h = lrelu(src2 @ W1 + b1) -> bf16
    gemm_kernel<2, false, false, true, false><<<GEMM_BLOCKS, blk, 0, stream>>>(
        s2bf, wt1, b1 + l * D_, nullptr, nullptr, hbf, nullptr, ROWS, D_, D_);

    // act = src + (h @ W2 + b2); fused bf16 store feeds next layer's GEMMs
    gemm_kernel<0, true, true, true, false><<<GEMM_BLOCKS, blk, 0, stream>>>(
        hbf, wt2, b2 + l * D_, src, act, Abf, nullptr, ROWS, D_, D_);
  }

  // final LayerNorm -> d_out (f32)
  layernorm_kernel<false><<<ROWW_BLOCKS, blk, 0, stream>>>(act, gf, bfv, outp, nullptr);
}
